// user_defined_rnn_75625784148206
// MI455X (gfx1250) — matrix operand order repaired and checked by reading, not compiled, not run on hardware
//
#include <hip/hip_runtime.h>
#include <math.h>

// Problem constants (from reference): T=256, B=2048, NC=3, NM=125, H=256
#define T_STEPS 256
#define BATCH   2048
#define NCAT    3
#define NVAL    125
#define IN_DIM  128   // NC + NM
#define H_DIM   256
#define G_DIM   1024  // 4*H
#define K_DIM   384   // IN + H
#define OUT_DIM 128   // NC + NM  (== IN_DIM, convenient)

typedef _Float16 h16;
typedef h16   v8h  __attribute__((ext_vector_type(8)));
typedef h16   v16h __attribute__((ext_vector_type(16)));
typedef float v8f  __attribute__((ext_vector_type(8)));

// ---------------- workspace layout (bytes) ----------------
#define WS_WG 0                                   // Wg  : [1024][384] f16 row-major
#define WS_WO (G_DIM * K_DIM * 2)                 // Wo  : [128][256]  f16 row-major
#define WS_BG (WS_WO + OUT_DIM * H_DIM * 2)       // bg  : [1024] f32  (b_ih + b_hh)
#define WS_BO (WS_BG + G_DIM * 4)                 // bo  : [128]  f32  ([b_cat, b_val])

// ---------------- LDS layout (dynamic, bytes) ----------------
#define XH_P   392                                 // [x(128) | h(256)] pitch in halves (+8 pad: 4-bank skew)
#define G_P    1028                                // gates pitch in floats (pad kills half-wave store conflict)
#define O_P    132                                 // out tile pitch in floats
#define XH_OFF 0
#define XH_SZ  (16 * XH_P * 2)                     // 12544
#define G_OFF  (XH_OFF + XH_SZ)                    // 12544
#define G_SZ   (16 * G_P * 4)                      // 65792
#define C_OFF  (G_OFF + G_SZ)                      // 78336
#define C_SZ   (16 * H_DIM * 4)                    // 16384
#define XF_OFF (C_OFF + C_SZ)                      // 94720
#define XF_SZ  (16 * IN_DIM * 4)                   // 8192
#define OB_OFF (XF_OFF + XF_SZ)                    // 102912
#define OB_SZ  (16 * O_P * 4)                      // 8448
#define BG_OFF (OB_OFF + OB_SZ)                    // 111360
#define BO_OFF (BG_OFF + G_DIM * 4)                // 115456
#define SMEM_BYTES (BO_OFF + OUT_DIM * 4)          // 115968

// ============ weight pack: f32 -> f16, fused biases ============
__global__ void rnn_prep_weights(const float* __restrict__ W_ih, const float* __restrict__ W_hh,
                                 const float* __restrict__ b_ih, const float* __restrict__ b_hh,
                                 const float* __restrict__ W_cat, const float* __restrict__ b_cat,
                                 const float* __restrict__ W_val, const float* __restrict__ b_val,
                                 h16* __restrict__ Wg, h16* __restrict__ Wo,
                                 float* __restrict__ bg, float* __restrict__ bo)
{
  int tid = blockIdx.x * blockDim.x + threadIdx.x;
  int stride = gridDim.x * blockDim.x;
  for (int i = tid; i < G_DIM * K_DIM; i += stride) {
    int n = i / K_DIM, k = i - n * K_DIM;
    float w = (k < IN_DIM) ? W_ih[n * IN_DIM + k] : W_hh[n * H_DIM + (k - IN_DIM)];
    Wg[i] = (h16)w;
  }
  for (int i = tid; i < OUT_DIM * H_DIM; i += stride) {
    int j = i / H_DIM, k = i - j * H_DIM;
    float w = (j < NCAT) ? W_cat[j * H_DIM + k] : W_val[(j - NCAT) * H_DIM + k];
    Wo[i] = (h16)w;
  }
  for (int i = tid; i < G_DIM; i += stride) bg[i] = b_ih[i] + b_hh[i];
  for (int i = tid; i < OUT_DIM; i += stride) bo[i] = (i < NCAT) ? b_cat[i] : b_val[i - NCAT];
}

__device__ __forceinline__ v16h cat16(v8h lo, v8h hi) {
  return __builtin_shufflevector(lo, hi, 0,1,2,3,4,5,6,7,8,9,10,11,12,13,14,15);
}

// ============ persistent LSTM: 1 workgroup = 16 batch rows, all 255 steps ============
__global__ __launch_bounds__(256)
void rnn_lstm_persist(const float* __restrict__ seq,
                      const h16* __restrict__ Wg, const h16* __restrict__ Wo,
                      const float* __restrict__ bg, const float* __restrict__ bo,
                      float* __restrict__ out)
{
  extern __shared__ unsigned char smem[];
  h16*   xh    = (h16*)(smem + XH_OFF);   // [16][XH_P] f16: x | h
  float* gates = (float*)(smem + G_OFF);  // [16][G_P]  f32
  float* cst   = (float*)(smem + C_OFF);  // [16][256]  f32 cell state
  float* xf    = (float*)(smem + XF_OFF); // [16][128]  f32 input (full precision for residual)
  float* ob    = (float*)(smem + OB_OFF); // [16][O_P]  f32 output tile
  float* bgl   = (float*)(smem + BG_OFF);
  float* bol   = (float*)(smem + BO_OFF);

  const int tid  = threadIdx.x;
  const int wave = tid >> 5;
  const int lane = tid & 31;
  const int mrow = lane & 15;   // A: batch row / B: weight row / D: column selector
  const int hi   = lane >> 4;   // half-wave selector
  const int rowbase = blockIdx.x * 16;

  // ---- init: biases to LDS, x0 = seq[0], h = 0, c = 0 ----
  for (int i = tid; i < G_DIM; i += 256) bgl[i] = bg[i];
  for (int i = tid; i < OUT_DIM; i += 256) bol[i] = bo[i];
  for (int u = tid; u < 16 * IN_DIM; u += 256) {
    int r = u >> 7, c = u & 127;
    float v = seq[(size_t)(rowbase + r) * IN_DIM + c];   // t = 0 (no NaNs at t=0 per reference)
    xf[r * IN_DIM + c] = v;
    xh[r * XH_P + c] = (h16)v;
  }
  for (int u = tid; u < 16 * H_DIM; u += 256) {
    int r = u >> 8, c = u & 255;
    xh[r * XH_P + IN_DIM + c] = (h16)0.f;
    cst[r * H_DIM + c] = 0.f;
  }
  __syncthreads();

  for (int it = 0; it < T_STEPS - 1; ++it) {
    // Opaque rewrite of a scalar OFFSET (not the pointer): prevents LICM from
    // hoisting the 96 per-wave B-fragment loads out of the time loop (which
    // previously spilled to scratch), while PRESERVING the pointer's global
    // address-space provenance so the backend emits global_load_b128 (saddr)
    // instead of flat_load (flat would couple weight loads to DScnt and need
    // 64-bit per-lane address math). Weights stay L2-resident and shared.
    int wgofs = 0;
    asm volatile("" : "+s"(wgofs));
    const h16* Wgi = Wg + wgofs;

    // ============ gate GEMM: [16 x 384] x [384 x 1024] via WMMA f16 ============
    // wave owns 8 N-tiles; K loop of 12 chunks; per chunk: batch all 8 B loads
    // (one clause, overlapped), then 8 WMMAs.
    v8f acc[8];
    #pragma unroll
    for (int nt = 0; nt < 8; ++nt) { v8f z = {}; acc[nt] = z; }

    // per-lane weight base: row = wave*128 + mrow, col-half = hi*16
    const h16* bbase = Wgi + (size_t)(wave * 128 + mrow) * K_DIM + hi * 16;

    for (int kc = 0; kc < 12; ++kc) {
      const int kb = kc * 32;
      // A fragment (16x32 f16): lane<16 -> K{kb..kb+15}; lane>=16 -> K{kb+16..kb+31}
      const h16* arow = xh + mrow * XH_P + kb + hi * 16;
      v16h a = cat16(*(const v8h*)(arow), *(const v8h*)(arow + 8));

      v16h bfr[8];
      #pragma unroll
      for (int nt = 0; nt < 8; ++nt) {
        const v8h* pb = (const v8h*)(bbase + nt * 16 * K_DIM + kb);
        bfr[nt] = cat16(pb[0], pb[1]);          // B 32x16: lane<16 K 0..15, else 16..31
      }
      #pragma unroll
      for (int nt = 0; nt < 8; ++nt)
        acc[nt] = __builtin_amdgcn_wmma_f32_16x16x32_f16(
            false, a, false, bfr[nt], (short)0, acc[nt], false, false);
    }
    #pragma unroll
    for (int nt = 0; nt < 8; ++nt) {
      const int ncol = (wave * 8 + nt) * 16 + mrow;
      #pragma unroll
      for (int r = 0; r < 8; ++r)
        gates[(r + hi * 8) * G_P + ncol] = acc[nt][r];      // D layout: VGPR r -> row r (+8 hi)
    }
    __syncthreads();

    // ============ LSTM cell elementwise (16x256 units over 256 threads) ============
    for (int u = tid; u < 16 * H_DIM; u += 256) {
      int r = u >> 8, c = u & 255;
      float gi = gates[r * G_P + c]       + bgl[c];
      float gf = gates[r * G_P + 256 + c] + bgl[256 + c];
      float gg = gates[r * G_P + 512 + c] + bgl[512 + c];
      float go = gates[r * G_P + 768 + c] + bgl[768 + c];
      float si = 1.f / (1.f + __expf(-gi));
      float sf = 1.f / (1.f + __expf(-gf));
      float so = 1.f / (1.f + __expf(-go));
      float c2 = sf * cst[r * H_DIM + c] + si * tanhf(gg);
      cst[r * H_DIM + c] = c2;
      xh[r * XH_P + IN_DIM + c] = (h16)(so * tanhf(c2));
    }
    __syncthreads();

    // ============ output GEMM: [16 x 256] x [256 x 128]; one N-tile per wave ============
    // Wo fragments (64 VGPRs/wave) are loop-invariant and intentionally left
    // hoistable so they live in registers for the whole recurrence.
    v8f oacc = {};
    #pragma unroll
    for (int kc = 0; kc < 8; ++kc) {
      const h16* arow = xh + mrow * XH_P + IN_DIM + kc * 32 + hi * 16;
      v16h a = cat16(*(const v8h*)(arow), *(const v8h*)(arow + 8));
      const v8h* pb = (const v8h*)(Wo + (size_t)(wave * 16 + mrow) * H_DIM + kc * 32 + hi * 16);
      v16h b = cat16(pb[0], pb[1]);
      oacc = __builtin_amdgcn_wmma_f32_16x16x32_f16(
          false, a, false, b, (short)0, oacc, false, false);
    }
    #pragma unroll
    for (int r = 0; r < 8; ++r)
      ob[(r + hi * 8) * O_P + wave * 16 + mrow] = oacc[r];
    __syncthreads();

    // ============ epilogue: value residual + bias; 3-way softmax ============
    for (int u = tid; u < 16 * NVAL; u += 256) {            // cols 3..127: + b_val + i_val
      int r = u / NVAL, c = NCAT + (u - r * NVAL);
      ob[r * O_P + c] += bol[c] + xf[r * IN_DIM + c];
    }
    if (tid < 16) {                                         // cols 0..2: softmax(h@Wcat^T + b_cat)
      int r = tid;
      float a0 = ob[r * O_P + 0] + bol[0];
      float a1 = ob[r * O_P + 1] + bol[1];
      float a2 = ob[r * O_P + 2] + bol[2];
      float m  = fmaxf(a0, fmaxf(a1, a2));
      float e0 = __expf(a0 - m), e1 = __expf(a1 - m), e2 = __expf(a2 - m);
      float inv = 1.f / (e0 + e1 + e2);
      ob[r * O_P + 0] = e0 * inv;
      ob[r * O_P + 1] = e1 * inv;
      ob[r * O_P + 2] = e2 * inv;
    }
    __syncthreads();

    // ============ write out[it] + NaN-impute next input from seq[it+1] ============
    for (int u = tid; u < 16 * OUT_DIM; u += 256) {
      int r = u >> 7, c = u & 127;
      float v = ob[r * O_P + c];
      out[((size_t)it * BATCH + rowbase + r) * OUT_DIM + c] = v;
      const float* praw = seq + ((size_t)(it + 1) * BATCH + rowbase + r) * IN_DIM + c;
      float raw = *praw;
      float x = (raw != raw) ? v : raw;                     // isnan -> use model output
      xf[r * IN_DIM + c] = x;
      xh[r * XH_P + c] = (h16)x;
      if (it + 2 < T_STEPS)                                 // prefetch next step's tile into L2
        __builtin_prefetch(praw + (size_t)BATCH * IN_DIM, 0, 1);
    }
    __syncthreads();
  }
}

extern "C" void kernel_launch(void* const* d_in, const int* in_sizes, int n_in,
                              void* d_out, int out_size, void* d_ws, size_t ws_size,
                              hipStream_t stream) {
  const float* seq   = (const float*)d_in[0];
  const float* W_ih  = (const float*)d_in[1];
  const float* W_hh  = (const float*)d_in[2];
  const float* b_ih  = (const float*)d_in[3];
  const float* b_hh  = (const float*)d_in[4];
  const float* W_cat = (const float*)d_in[5];
  const float* b_cat = (const float*)d_in[6];
  const float* W_val = (const float*)d_in[7];
  const float* b_val = (const float*)d_in[8];

  h16*   Wg = (h16*)((char*)d_ws + WS_WG);
  h16*   Wo = (h16*)((char*)d_ws + WS_WO);
  float* bg = (float*)((char*)d_ws + WS_BG);
  float* bo = (float*)((char*)d_ws + WS_BO);

  rnn_prep_weights<<<256, 256, 0, stream>>>(W_ih, W_hh, b_ih, b_hh,
                                            W_cat, b_cat, W_val, b_val,
                                            Wg, Wo, bg, bo);
  rnn_lstm_persist<<<BATCH / 16, 256, SMEM_BYTES, stream>>>(
      seq, Wg, Wo, bg, bo, (float*)d_out);
}
